// AngleLossV2_38800734552572
// MI455X (gfx1250) — compile-verified, matrix operand order checked by reading, not run on hardware
//
#include <hip/hip_runtime.h>

#define Nn 384
#define Dd 128
#define STRIDE 136          // 128 + 8 bf16 pad -> row stride 272B, kills LDS bank conflicts
#define NORM_EPS 1e-6f
#define PD_EPS   1e-6f

typedef __attribute__((ext_vector_type(16))) __bf16 v16bf;
typedef __attribute__((ext_vector_type(4)))  __bf16 v4bf;
typedef __attribute__((ext_vector_type(8)))  float  v8f;

__device__ __forceinline__ float wave_reduce(float v) {
#pragma unroll
  for (int off = 16; off > 0; off >>= 1) v += __shfl_xor(v, off, 32);
  return v;
}

// Fragment for 16x16x32 bf16 WMMA: lane L(0-15): row=base+L, K = k0..k0+7 and k0+16..k0+23
// lane L+16: row=base+L, K = k0+8..k0+15 and k0+24..k0+31. Two ds_load_b128 per lane.
// For Gram (B = A^T of same matrix) the B fragment uses the identical load pattern.
__device__ __forceinline__ v16bf frag_load(const __bf16* base, int rowBase, int k0, int lane) {
  int row  = rowBase + (lane & 15);
  int koff = k0 + ((lane >> 4) << 3);
  const __bf16* p = base + row * STRIDE + koff;
  v16bf f;
  ((uint4*)&f)[0] = *(const uint4*)(p);
  ((uint4*)&f)[1] = *(const uint4*)(p + 16);
  return f;
}

#define WMMA_BF16(A, B, C) \
  __builtin_amdgcn_wmma_f32_16x16x32_bf16(false, (A), false, (B), (short)0, (C), false, false)

__global__ void init_ws_kernel(float* ws) {
  if (threadIdx.x < 8) ws[threadIdx.x] = 0.0f;
}

// PASS 1: X = true  -> accumulate ws[0]=cnt, ws[1]=s1, ws[2]=s2
// PASS 2: X = feat  -> accumulate ws[3]=sum of masked distances
template <int PASS>
__global__ __launch_bounds__(256) void gram_kernel(const float* __restrict__ X,
                                                   const int* __restrict__ pm,
                                                   float* __restrict__ ws) {
  extern __shared__ char smem[];
  __bf16* hiA = (__bf16*)smem;                    // 384 x 136 bf16 (hi part)
  __bf16* loA = hiA + Nn * STRIDE;                // 384 x 136 bf16 (lo residual)
  float*  mw  = (float*)(loA + Nn * STRIDE);      // 384 mask weights (0.0 / 1.0)
  __shared__ int Mcount;

  const int a    = blockIdx.x;
  const int tid  = threadIdx.x;
  const int lane = tid & 31;
  const int wave = tid >> 5;

  if (tid == 0) Mcount = 0;
  __syncthreads();

  // ---- per-a mask weights: m[p] = pm[a,p] && p != a ----
  int localM = 0;
  for (int p = tid; p < Nn; p += 256) {
    int f = (pm[(size_t)a * Nn + p] != 0) && (p != a);
    mw[p] = (float)f;
    localM += f;
  }
  atomicAdd(&Mcount, localM);

  // ---- load rows, L2-normalize, split into bf16 hi + lo residual ----
  for (int r = wave; r < Nn; r += 8) {
    const float4 v = *(const float4*)(X + ((size_t)a * Nn + r) * Dd + lane * 4);
    float ss = v.x * v.x + v.y * v.y + v.z * v.z + v.w * v.w;
    ss = wave_reduce(ss);
    float scale = 1.0f / fmaxf(sqrtf(ss), NORM_EPS);
    float e[4] = {v.x * scale, v.y * scale, v.z * scale, v.w * scale};
    v4bf hv, lv;
#pragma unroll
    for (int q = 0; q < 4; ++q) {
      __bf16 h = (__bf16)e[q];
      hv[q] = h;
      lv[q] = (__bf16)(e[q] - (float)h);
    }
    *(v4bf*)(hiA + r * STRIDE + lane * 4) = hv;
    *(v4bf*)(loA + r * STRIDE + lane * 4) = lv;
  }
  __syncthreads();

  if (PASS == 1 && tid == 0) {
    float M = (float)Mcount;                 // cnt contribution = M*(M-1) (p!=q pairs)
    atomicAdd(&ws[0], M * (M - 1.0f));
  }

  float cnt = 0.f, s1 = 0.f, s2 = 0.f;
  if (PASS == 2) { cnt = ws[0]; s1 = ws[1]; s2 = ws[2]; }

  float acc0 = 0.0f, acc1 = 0.0f;           // (s1,s2) or (sum_d, unused), half-weighted

  // Branchless per-16x16-tile epilogue: weight = mw[p]*mw[q]*(q>p)
  auto epilogue = [&](const v8f& accv, int pbase, int qbase) {
    const int q = qbase + (lane & 15);
    const float wq = mw[q];
    const int prow = pbase + ((lane >> 4) << 3);
#pragma unroll
    for (int v = 0; v < 8; ++v) {
      const int p = prow + v;
      float w = wq * mw[p];
      w = (q > p) ? w : 0.0f;
      const float g = accv[v];
      if (PASS == 1) {
        const float tv = g - PD_EPS;
        acc0 += w * tv;
        acc1 += w * tv * tv;
      } else {
        const float d2 = fmaf(cnt * g, g, fmaf(-2.0f * s1, g, s2));
        acc0 += w * sqrtf(fmaxf(d2, 0.0f));
      }
    }
  };

  // ---- Gram via bf16x3 compensated WMMA ----
  // 2x2 register blocking: each wave computes a 32x32 output block (4 accumulators,
  // 12 WMMAs per k-step from 8 fragments -> 2x LDS reuse vs 1 tile).
  // Upper-triangular 32x32 blocks only: 78 of 144; q>p weight discards the
  // below-diagonal sub-tile of the 12 diagonal blocks.
  for (int t = wave; t < 78; t += 8) {
    int bi = 0, rem = t;
    while (rem >= 12 - bi) { rem -= 12 - bi; ++bi; }
    const int bj = bi + rem;
    const int p0 = bi * 32, q0 = bj * 32;

    v8f acc00 = {}, acc01 = {}, acc10 = {}, acc11 = {};
#pragma unroll
    for (int kk = 0; kk < 4; ++kk) {
      const int k0 = kk * 32;
      v16bf a0h = frag_load(hiA, p0,      k0, lane);
      v16bf a0l = frag_load(loA, p0,      k0, lane);
      v16bf a1h = frag_load(hiA, p0 + 16, k0, lane);
      v16bf a1l = frag_load(loA, p0 + 16, k0, lane);
      v16bf b0h = frag_load(hiA, q0,      k0, lane);
      v16bf b0l = frag_load(loA, q0,      k0, lane);
      v16bf b1h = frag_load(hiA, q0 + 16, k0, lane);
      v16bf b1l = frag_load(loA, q0 + 16, k0, lane);

      acc00 = WMMA_BF16(a0h, b0h, acc00);
      acc01 = WMMA_BF16(a0h, b1h, acc01);
      acc10 = WMMA_BF16(a1h, b0h, acc10);
      acc11 = WMMA_BF16(a1h, b1h, acc11);
      acc00 = WMMA_BF16(a0h, b0l, acc00);
      acc01 = WMMA_BF16(a0h, b1l, acc01);
      acc10 = WMMA_BF16(a1h, b0l, acc10);
      acc11 = WMMA_BF16(a1h, b1l, acc11);
      acc00 = WMMA_BF16(a0l, b0h, acc00);
      acc01 = WMMA_BF16(a0l, b1h, acc01);
      acc10 = WMMA_BF16(a1l, b0h, acc10);
      acc11 = WMMA_BF16(a1l, b1h, acc11);
    }

    epilogue(acc00, p0,      q0);
    epilogue(acc01, p0,      q0 + 16);
    epilogue(acc10, p0 + 16, q0);
    epilogue(acc11, p0 + 16, q0 + 16);
  }

  // symmetry weight 2 folded in here
  acc0 = wave_reduce(acc0);
  if (PASS == 1) {
    acc1 = wave_reduce(acc1);
    if (lane == 0) { atomicAdd(&ws[1], 2.0f * acc0); atomicAdd(&ws[2], 2.0f * acc1); }
  } else {
    if (lane == 0) atomicAdd(&ws[3], 2.0f * acc0);
  }
}

__global__ void finalize_kernel(const float* __restrict__ ws, float* __restrict__ out) {
  out[0] = 0.5f * ws[3] / fmaxf(ws[0], 1.0f);
}

extern "C" void kernel_launch(void* const* d_in, const int* in_sizes, int n_in,
                              void* d_out, int out_size, void* d_ws, size_t ws_size,
                              hipStream_t stream) {
  (void)in_sizes; (void)n_in; (void)out_size; (void)ws_size;
  const float* feat = (const float*)d_in[0];
  const int*   pm   = (const int*)d_in[1];
  const float* tru  = (const float*)d_in[2];
  float* ws  = (float*)d_ws;
  float* out = (float*)d_out;

  // 2*384*136*2B (bf16 hi+lo) + 384*4B mask weights  ~= 206 KB of 320 KB WGP LDS
  constexpr size_t SMEM = (size_t)2 * Nn * STRIDE * sizeof(__bf16) + Nn * sizeof(float) + 256;

  // Allow >64KB dynamic LDS (idempotent, deterministic; not a stream op).
  hipFuncSetAttribute(reinterpret_cast<const void*>(gram_kernel<1>),
                      hipFuncAttributeMaxDynamicSharedMemorySize, (int)SMEM);
  hipFuncSetAttribute(reinterpret_cast<const void*>(gram_kernel<2>),
                      hipFuncAttributeMaxDynamicSharedMemorySize, (int)SMEM);

  init_ws_kernel<<<1, 32, 0, stream>>>(ws);
  gram_kernel<1><<<dim3(Nn), dim3(256), SMEM, stream>>>(tru, pm, ws);   // t stats
  gram_kernel<2><<<dim3(Nn), dim3(256), SMEM, stream>>>(feat, pm, ws);  // loss sum
  finalize_kernel<<<1, 1, 0, stream>>>(ws, out);
}
